// Mamba2Mixer_7954279432300
// MI455X (gfx1250) — matrix operand order repaired and checked by reading, not compiled, not run on hardware
//
#include <hip/hip_runtime.h>
#include <stdint.h>

// ---------------- problem constants ----------------
#define HID    2048
#define DSTATE 128
#define KCONV  4
#define DIN    4096
#define HDIM   64
#define NH     64
#define CHUNK  256
#define CONVD  (DIN + 2*DSTATE)          // 4352
#define PROJ   (2*(DIN+DSTATE)+NH)       // 8512
#define NPAD   8576                      // 67*128, padded proj width for clean tiling
#define BATCH  2
#define SEQ    4096
#define MTOK   (BATCH*SEQ)               // 8192
#define NCHUNK (SEQ/CHUNK)               // 16
#define NBLK   (BATCH*NCHUNK*NH)         // 2048 (b,c,h) tiles
#define EPSF   1e-5f

typedef __bf16 bf16;
typedef __bf16        v16bf __attribute__((ext_vector_type(16)));
typedef float         v8f   __attribute__((ext_vector_type(8)));
typedef unsigned int  u32x4 __attribute__((ext_vector_type(4)));

union FragU { v16bf v; u32x4 q[2]; };

// A fragment (16x32, 16-bit, ISA 7.12.2): lane r=lane&15 holds row M=r,
// half hf=lane>>4 selects K elements {hf*8..hf*8+7} and {16+hf*8..23+hf*8}.
__device__ __forceinline__ v16bf load_a_frag(const bf16* row_k0, int hf) {
  FragU u;
  u.q[0] = *(const u32x4*)(row_k0 + hf*8);
  u.q[1] = *(const u32x4*)(row_k0 + 16 + hf*8);
  return u.v;
}
// B fragment (32x16, 16-bit): lane r holds column N=r; half hf holds
// K elements {hf*8..hf*8+7} and {16+hf*8..23+hf*8} (same order as A).
__device__ __forceinline__ v16bf load_b_frag(const bf16* col_k0, int hf) {
  FragU u;
  u.q[0] = *(const u32x4*)(col_k0 + hf*8);
  u.q[1] = *(const u32x4*)(col_k0 + 16 + hf*8);
  return u.v;
}

__device__ __forceinline__ v8f wmma_bf16(v16bf a, v16bf b, v8f c) {
  return __builtin_amdgcn_wmma_f32_16x16x32_bf16(false, a, false, b,
                                                 (short)0, c, false, false);
}

// Async global->LDS copy of 16B (ASYNCcnt tracked, CDNA5 ISA).
// lds_off is the byte offset within the wave's LDS allocation: low 32 bits of
// the generic shared-memory address (flat aperture rule, ISA 10.2).
__device__ __forceinline__ void async_cp16(uint32_t lds_off, const void* g) {
  asm volatile("global_load_async_to_lds_b128 %0, %1, off"
               :: "v"(lds_off), "v"((unsigned long long)(uintptr_t)g)
               : "memory");
}
__device__ __forceinline__ void wait_async0() {
  asm volatile("s_wait_asynccnt 0x0" ::: "memory");
}

__device__ __forceinline__ float siluf(float x) {
  return x / (1.0f + __expf(-x));
}

// ---------------- casts ----------------
__global__ void k_cast_bf16(const float* __restrict__ in, bf16* __restrict__ out, long n) {
  long i = (long)blockIdx.x * blockDim.x + threadIdx.x;
  long stride = (long)gridDim.x * blockDim.x;
  for (; i < n; i += stride) out[i] = (bf16)in[i];
}

__global__ void k_cast_pad_w(const float* __restrict__ in, bf16* __restrict__ out) {
  // in: [PROJ, HID] f32 -> out: [NPAD, HID] bf16, rows >= PROJ zeroed
  const long n = (long)NPAD * HID;
  long i = (long)blockIdx.x * blockDim.x + threadIdx.x;
  long stride = (long)gridDim.x * blockDim.x;
  for (; i < n; i += stride) {
    long row = i / HID;
    out[i] = (row < PROJ) ? (bf16)in[i] : (bf16)0.0f;
  }
}

// ---------------- bf16 WMMA GEMM: C[M,N] = A[M,K] * B[N,K]^T ----------------
// block tile 128x128, 8 waves arranged 2(M) x 4(N), each wave 64x32 (4x2 tiles).
// A/B K-slabs (128x32 bf16 = 8KB each) are double-buffered in LDS and filled
// with GLOBAL_LOAD_ASYNC_TO_LDS_B128, overlapping the WMMAs of the prior slab.
__global__ void k_gemm_bf16(const bf16* __restrict__ A, const bf16* __restrict__ B,
                            float* __restrict__ C, int Kdim, int ldc) {
  extern __shared__ char smem[];
  bf16* sA = (bf16*)smem;               // [2][128][32]
  bf16* sB = sA + 2 * 128 * 32;         // [2][128][32]
  const int tid  = threadIdx.x;
  const int w    = tid >> 5, lane = tid & 31;
  const int r    = lane & 15, hf = lane >> 4;
  const long bm  = (long)blockIdx.y * 128;
  const long bn  = (long)blockIdx.x * 128;
  const int wml  = (w >> 2) * 64;       // wave M offset inside block
  const int wnl  = (w & 3) * 32;        // wave N offset inside block

  // staging map: 512 x 16B transfers per tile; thread covers t and t+256.
  const int t0 = tid, t1 = tid + 256;
  const int row0 = t0 >> 2, seg0 = t0 & 3;      // row is 64B = 4 segments
  const int row1 = t1 >> 2, seg1 = t1 & 3;
  const uint32_t ldsA = (uint32_t)(uintptr_t)sA;
  const uint32_t ldsB = (uint32_t)(uintptr_t)sB;

  auto issue = [&](int buf, int k0) {
    const uint32_t bo = (uint32_t)buf * 128 * 32 * 2;   // bytes
    async_cp16(ldsA + bo + (uint32_t)(row0 * 32 + seg0 * 8) * 2,
               A + (bm + row0) * (long)Kdim + k0 + seg0 * 8);
    async_cp16(ldsA + bo + (uint32_t)(row1 * 32 + seg1 * 8) * 2,
               A + (bm + row1) * (long)Kdim + k0 + seg1 * 8);
    async_cp16(ldsB + bo + (uint32_t)(row0 * 32 + seg0 * 8) * 2,
               B + (bn + row0) * (long)Kdim + k0 + seg0 * 8);
    async_cp16(ldsB + bo + (uint32_t)(row1 * 32 + seg1 * 8) * 2,
               B + (bn + row1) * (long)Kdim + k0 + seg1 * 8);
  };

  v8f acc[4][2] = {};
  const int nk = Kdim / 32;
  issue(0, 0);
  for (int kt = 0; kt < nk; ++kt) {
    const int buf = kt & 1;
    wait_async0();        // own slice of buf resident
    __syncthreads();      // everyone's slice resident; prior reads of buf^1 done
    if (kt + 1 < nk) issue(buf ^ 1, (kt + 1) * 32);
    const bf16* tA = sA + buf * 128 * 32;
    const bf16* tB = sB + buf * 128 * 32;
    v16bf a[4], b2[2];
#pragma unroll
    for (int i = 0; i < 4; i++)
      a[i] = load_a_frag(tA + (wml + i*16 + r) * 32, hf);
#pragma unroll
    for (int j = 0; j < 2; j++)
      b2[j] = load_b_frag(tB + (wnl + j*16 + r) * 32, hf);
#pragma unroll
    for (int i = 0; i < 4; i++)
#pragma unroll
      for (int j = 0; j < 2; j++)
        acc[i][j] = wmma_bf16(a[i], b2[j], acc[i][j]);
  }
#pragma unroll
  for (int i = 0; i < 4; i++)
#pragma unroll
    for (int j = 0; j < 2; j++)
#pragma unroll
      for (int jj = 0; jj < 8; jj++)
        C[(bm + wml + i*16 + jj + 8*hf) * (long)ldc + bn + wnl + j*16 + r] = acc[i][j][jj];
}

// ---------------- dt: softplus, A*dt, per-chunk cumsum ----------------
__global__ void k_dt(const float* __restrict__ zx, const float* __restrict__ dt_bias,
                     const float* __restrict__ A_log, float* __restrict__ dt_sp,
                     float* __restrict__ Acs, float* __restrict__ chunk_sum) {
  const int blk = blockIdx.x;               // ((b*NC + c)*NH + h)
  const int h  = blk % NH;
  const int bc = blk / NH;
  const int c  = bc % NCHUNK, b = bc / NCHUNK;
  const int l  = threadIdx.x;
  const long tok = (long)b * SEQ + c * CHUNK + l;
  float v  = zx[tok * NPAD + DIN + CONVD + h] + dt_bias[h];
  float sp = (v > 20.0f) ? v : log1pf(__expf(v));
  float Ah = -__expf(A_log[h]);
  dt_sp[tok * NH + h] = sp;
  __shared__ float buf[CHUNK];
  buf[l] = sp * Ah;
  __syncthreads();
  for (int off = 1; off < CHUNK; off <<= 1) {   // inclusive Hillis-Steele scan
    float t = (l >= off) ? buf[l - off] : 0.0f;
    __syncthreads();
    buf[l] += t;
    __syncthreads();
  }
  float cs = buf[l];
  Acs[(long)blk * CHUNK + l] = cs;
  if (l == CHUNK - 1) chunk_sum[blk] = cs;
}

// ---------------- causal conv1d (K=4) + silu, split + bf16 pack ----------------
__global__ void k_conv(const float* __restrict__ zx, const float* __restrict__ cw,
                       const float* __restrict__ cb, const float* __restrict__ dt_sp,
                       bf16* __restrict__ xb, bf16* __restrict__ xdtb,
                       bf16* __restrict__ Bb, bf16* __restrict__ Cb) {
  long gid = (long)blockIdx.x * blockDim.x + threadIdx.x;
  if (gid >= (long)MTOK * CONVD) return;
  const int  ch  = (int)(gid % CONVD);
  const long tok = gid / CONVD;
  const int  b   = (int)(tok / SEQ), l = (int)(tok % SEQ);
  float acc = cb[ch];
#pragma unroll
  for (int i = 0; i < KCONV; i++) {
    int li = l - (KCONV - 1) + i;
    if (li >= 0)
      acc += zx[((long)b * SEQ + li) * NPAD + DIN + ch] * cw[ch * KCONV + i];
  }
  float y = siluf(acc);
  if (ch < DIN) {
    int h = ch >> 6;
    xb[tok * DIN + ch]   = (bf16)y;
    xdtb[tok * DIN + ch] = (bf16)(y * dt_sp[tok * NH + h]);
  } else if (ch < DIN + DSTATE) {
    Bb[tok * DSTATE + (ch - DIN)] = (bf16)y;
  } else {
    Cb[tok * DSTATE + (ch - DIN - DSTATE)] = (bf16)y;
  }
}

// ---------------- chunk_states[p,n] = sum_l xdt[l,p]*decay[l]*B[l,n] ----------------
__global__ void k_chunk_states(const bf16* __restrict__ xdtb, const bf16* __restrict__ Bb,
                               const float* __restrict__ Acs, const float* __restrict__ csum,
                               float* __restrict__ states) {
  extern __shared__ char smem[];
  const int LDT = CHUNK + 8;
  bf16* xT  = (bf16*)smem;              // [HDIM][LDT]
  bf16* BrT = xT + HDIM * LDT;          // [DSTATE][LDT]
  const int blk = blockIdx.x;
  const int h  = blk % NH;
  const int bc = blk / NH;
  const int c  = bc % NCHUNK, b = bc / NCHUNK;
  const long tok0 = (long)b * SEQ + c * CHUNK;
  const int l = threadIdx.x;
  float dec = __expf(csum[blk] - Acs[(long)blk * CHUNK + l]);
  const bf16* xr = xdtb + (tok0 + l) * DIN + h * HDIM;
  for (int p = 0; p < HDIM; p++) xT[p * LDT + l] = (bf16)((float)xr[p] * dec);
  const bf16* br = Bb + (tok0 + l) * DSTATE;
  for (int n = 0; n < DSTATE; n++) BrT[n * LDT + l] = br[n];
  __syncthreads();
  const int w = threadIdx.x >> 5, lane = threadIdx.x & 31;
  const int r = lane & 15, hf = lane >> 4;
  const int pt = w & 3;        // p tile
  const int ng = w >> 2;       // n tile group (4 tiles each)
  v8f acc[4] = {};
  for (int k0 = 0; k0 < CHUNK; k0 += 32) {
    v16bf a = load_a_frag(xT + (pt*16 + r) * LDT + k0, hf);
#pragma unroll
    for (int j = 0; j < 4; j++) {
      v16bf bfr = load_b_frag(BrT + ((ng*4 + j)*16 + r) * LDT + k0, hf);
      acc[j] = wmma_bf16(a, bfr, acc[j]);
    }
  }
  float* out = states + (long)blk * HDIM * DSTATE;
#pragma unroll
  for (int j = 0; j < 4; j++)
#pragma unroll
    for (int jj = 0; jj < 8; jj++)
      out[(pt*16 + jj + 8*hf) * DSTATE + (ng*4 + j)*16 + r] = acc[j][jj];
}

// ---------------- inter-chunk sequential scan ----------------
__global__ void k_scan(const float* __restrict__ states, const float* __restrict__ csum,
                       float* __restrict__ prev) {
  const int bh = blockIdx.x;       // b*NH + h
  const int h = bh % NH, b = bh / NH;
  const int t = threadIdx.x;
  float run[32];
#pragma unroll
  for (int i = 0; i < 32; i++) run[i] = 0.0f;
  for (int c = 0; c < NCHUNK; c++) {
    const int blk = (b * NCHUNK + c) * NH + h;
    const long base = (long)blk * HDIM * DSTATE + (long)t * 32;
#pragma unroll
    for (int i = 0; i < 32; i++) prev[base + i] = run[i];
    float dec = __expf(csum[blk]);
#pragma unroll
    for (int i = 0; i < 32; i++) run[i] = run[i] * dec + states[base + i];
  }
}

// ---------------- Y = (C B^T . Lmat) xdt + exp(Acs) (C prev^T) + D*x ----------------
__global__ void k_ssd_y(const bf16* __restrict__ Cb, const bf16* __restrict__ Bb,
                        const bf16* __restrict__ xdtb, const bf16* __restrict__ xb,
                        const float* __restrict__ Acs, const float* __restrict__ prev,
                        const float* __restrict__ Dv, float* __restrict__ Y) {
  extern __shared__ char smem[];
  const int LDT = CHUNK + 8;
  const int LDP = DSTATE + 8;
  bf16*  xdtT  = (bf16*)smem;                        // [HDIM][LDT]
  bf16*  prevb = xdtT + HDIM * LDT;                  // [HDIM][LDP]
  bf16*  wbuf  = prevb + HDIM * LDP;                 // [8 waves][32][32]
  float* sAcs  = (float*)(wbuf + 8 * 32 * 32);       // [CHUNK]
  const int blk = blockIdx.x;
  const int h  = blk % NH;
  const int bc = blk / NH;
  const int c  = bc % NCHUNK, b = bc / NCHUNK;
  const long tok0 = (long)b * SEQ + c * CHUNK;
  const int tid = threadIdx.x;
  // stage: Acs, transposed xdt, bf16 prev
  sAcs[tid] = Acs[(long)blk * CHUNK + tid];
  {
    const bf16* xr = xdtb + (tok0 + tid) * DIN + h * HDIM;
    for (int p = 0; p < HDIM; p++) xdtT[p * LDT + tid] = xr[p];
    const float* pv = prev + (long)blk * HDIM * DSTATE;
    for (int i = 0; i < 32; i++) {
      int idx = tid * 32 + i;
      prevb[(idx >> 7) * LDP + (idx & 127)] = (bf16)pv[idx];
    }
  }
  __syncthreads();
  const int w = tid >> 5, lane = tid & 31;
  const int r = lane & 15, hf = lane >> 4;
  const int l0 = w * 32;                // this wave's 32 rows of the chunk
  bf16* wb = wbuf + w * 32 * 32;        // per-wave scratch tile
  v8f yacc[2][4] = {};
  // ---- intra-chunk (causal: wave w only needs s-blocks sb <= w) ----
  for (int sb = 0; sb <= w; sb++) {
    const int s0 = sb * 32;
    v8f g[2][2] = {};
    for (int kn = 0; kn < DSTATE; kn += 32) {
      v16bf aC[2], bB[2];
#pragma unroll
      for (int t = 0; t < 2; t++)
        aC[t] = load_a_frag(Cb + (tok0 + l0 + t*16 + r) * DSTATE + kn, hf);
#pragma unroll
      for (int st = 0; st < 2; st++)
        bB[st] = load_b_frag(Bb + (tok0 + s0 + st*16 + r) * DSTATE + kn, hf);
#pragma unroll
      for (int t = 0; t < 2; t++)
#pragma unroll
        for (int st = 0; st < 2; st++)
          g[t][st] = wmma_bf16(aC[t], bB[st], g[t][st]);
    }
    // mask + decay weights: W[l,s] = G * exp(Acs[l]-Acs[s]) for s<=l, repack via LDS
#pragma unroll
    for (int t = 0; t < 2; t++)
#pragma unroll
      for (int st = 0; st < 2; st++)
#pragma unroll
        for (int jj = 0; jj < 8; jj++) {
          int lr = t*16 + jj + 8*hf;
          int ll = l0 + lr;
          int ss = s0 + st*16 + r;
          float val = (ss <= ll) ? g[t][st][jj] * __expf(sAcs[ll] - sAcs[ss]) : 0.0f;
          wb[lr * 32 + st*16 + r] = (bf16)val;
        }
    // same-wave LDS RAW: compiler inserts dscnt waits before the ds reads
    v16bf aW0 = load_a_frag(wb + r * 32, hf);
    v16bf aW1 = load_a_frag(wb + (16 + r) * 32, hf);
#pragma unroll
    for (int pt = 0; pt < 4; pt++) {
      v16bf bX = load_b_frag(xdtT + (pt*16 + r) * LDT + s0, hf);
      yacc[0][pt] = wmma_bf16(aW0, bX, yacc[0][pt]);
      yacc[1][pt] = wmma_bf16(aW1, bX, yacc[1][pt]);
    }
  }
  // ---- inter-chunk: Yo = (exp(Acs[l]) * C[l,:]) . prev[p,:]^T ----
  for (int kn = 0; kn < DSTATE; kn += 32) {
    v16bf aC[2];
#pragma unroll
    for (int t = 0; t < 2; t++) {
      int row = l0 + t*16 + r;
      const bf16* src = Cb + (tok0 + row) * DSTATE + kn;
      FragU u;
      u.q[0] = *(const u32x4*)(src + hf*8);
      u.q[1] = *(const u32x4*)(src + 16 + hf*8);
      float esc = __expf(sAcs[row]);
#pragma unroll
      for (int e = 0; e < 16; e++) u.v[e] = (bf16)((float)u.v[e] * esc);
      aC[t] = u.v;
    }
#pragma unroll
    for (int pt = 0; pt < 4; pt++) {
      v16bf bP = load_b_frag(prevb + (pt*16 + r) * LDP + kn, hf);
      yacc[0][pt] = wmma_bf16(aC[0], bP, yacc[0][pt]);
      yacc[1][pt] = wmma_bf16(aC[1], bP, yacc[1][pt]);
    }
  }
  // ---- + D*x, store f32 ----
  const float Dh = Dv[h];
#pragma unroll
  for (int t = 0; t < 2; t++)
#pragma unroll
    for (int pt = 0; pt < 4; pt++)
#pragma unroll
      for (int jj = 0; jj < 8; jj++) {
        int ll = l0 + t*16 + jj + 8*hf;
        long tok = tok0 + ll;
        int p = pt*16 + r;
        float xv = (float)xb[tok * DIN + h * HDIM + p];
        Y[tok * DIN + h * HDIM + p] = yacc[t][pt][jj] + Dh * xv;
      }
}

// ---------------- gated RMSNorm -> bf16 ----------------
__global__ void k_norm(const float* __restrict__ Y, const float* __restrict__ zx,
                       const float* __restrict__ nw, bf16* __restrict__ yn) {
  const long row = blockIdx.x;
  const int t = threadIdx.x;
  float v[16];
  float ss = 0.0f;
#pragma unroll
  for (int i = 0; i < 16; i++) {
    int col = t * 16 + i;
    float y = Y[row * DIN + col];
    float z = zx[row * NPAD + col];
    float yv = y * siluf(z);
    v[i] = yv;
    ss += yv * yv;
  }
  __shared__ float red[256];
  red[t] = ss;
  __syncthreads();
  for (int off = 128; off > 0; off >>= 1) {
    if (t < off) red[t] += red[t + off];
    __syncthreads();
  }
  float inv = rsqrtf(red[0] / (float)DIN + EPSF);
#pragma unroll
  for (int i = 0; i < 16; i++) {
    int col = t * 16 + i;
    yn[row * DIN + col] = (bf16)(v[i] * inv * nw[col]);
  }
}

// ---------------- launch ----------------
extern "C" void kernel_launch(void* const* d_in, const int* in_sizes, int n_in,
                              void* d_out, int out_size, void* d_ws, size_t ws_size,
                              hipStream_t stream) {
  (void)in_sizes; (void)n_in; (void)out_size; (void)ws_size;
  const float* hs    = (const float*)d_in[0];
  const float* w_in  = (const float*)d_in[1];
  const float* convw = (const float*)d_in[2];
  const float* convb = (const float*)d_in[3];
  const float* dt_b  = (const float*)d_in[4];
  const float* A_log = (const float*)d_in[5];
  const float* Dv    = (const float*)d_in[6];
  const float* normw = (const float*)d_in[7];
  const float* w_out = (const float*)d_in[8];
  float* out = (float*)d_out;

  // workspace layout (~845 MB total)
  char* ws = (char*)d_ws;
  auto alloc = [&](size_t bytes) {
    char* p = ws;
    ws += (bytes + 255) & ~(size_t)255;
    return p;
  };
  bf16*  hs_b   = (bf16*) alloc((size_t)MTOK * HID * 2);
  bf16*  win_b  = (bf16*) alloc((size_t)NPAD * HID * 2);
  bf16*  wout_b = (bf16*) alloc((size_t)HID * DIN * 2);
  float* zx     = (float*)alloc((size_t)MTOK * NPAD * 4);
  float* dt_sp  = (float*)alloc((size_t)MTOK * NH * 4);
  float* AcsA   = (float*)alloc((size_t)NBLK * CHUNK * 4);
  float* csum   = (float*)alloc((size_t)NBLK * 4);
  bf16*  xb     = (bf16*) alloc((size_t)MTOK * DIN * 2);
  bf16*  xdtb   = (bf16*) alloc((size_t)MTOK * DIN * 2);
  bf16*  Bb     = (bf16*) alloc((size_t)MTOK * DSTATE * 2);
  bf16*  Cbm    = (bf16*) alloc((size_t)MTOK * DSTATE * 2);
  float* states = (float*)alloc((size_t)NBLK * HDIM * DSTATE * 4);
  float* prevB  = (float*)alloc((size_t)NBLK * HDIM * DSTATE * 4);
  float* Ybuf   = (float*)alloc((size_t)MTOK * DIN * 4);
  bf16*  yn     = (bf16*) alloc((size_t)MTOK * DIN * 2);

  // 1. casts to bf16
  k_cast_bf16<<<4096, 256, 0, stream>>>(hs, hs_b, (long)MTOK * HID);
  k_cast_pad_w<<<4096, 256, 0, stream>>>(w_in, win_b);
  k_cast_bf16<<<4096, 256, 0, stream>>>(w_out, wout_b, (long)HID * DIN);

  // 2. in_proj GEMM (async-LDS double buffered): zx = hs_b * win_b^T
  const size_t smg = (size_t)2 * 2 * 128 * 32 * sizeof(bf16);  // 32 KB
  dim3 g1(NPAD / 128, MTOK / 128);
  k_gemm_bf16<<<g1, 256, smg, stream>>>(hs_b, win_b, zx, HID, NPAD);

  // 3. dt softplus + A*dt cumsum per chunk
  k_dt<<<NBLK, CHUNK, 0, stream>>>(zx, dt_b, A_log, dt_sp, AcsA, csum);

  // 4. causal conv + silu + split
  long nconv = (long)MTOK * CONVD;
  k_conv<<<(unsigned)((nconv + 255) / 256), 256, 0, stream>>>(zx, convw, convb, dt_sp,
                                                              xb, xdtb, Bb, Cbm);

  // 5. per-chunk states GEMM (dyn LDS: transposed xdt + B)
  size_t sm5 = (size_t)(HDIM + DSTATE) * (CHUNK + 8) * sizeof(bf16);
  k_chunk_states<<<NBLK, 256, sm5, stream>>>(xdtb, Bb, AcsA, csum, states);

  // 6. inter-chunk scan
  k_scan<<<BATCH * NH, 256, 0, stream>>>(states, csum, prevB);

  // 7. Yd + Yo + D*x
  size_t sm7 = (size_t)HDIM * (CHUNK + 8) * 2 + (size_t)HDIM * (DSTATE + 8) * 2
             + (size_t)8 * 32 * 32 * 2 + (size_t)CHUNK * 4;
  k_ssd_y<<<NBLK, 256, sm7, stream>>>(Cbm, Bb, xdtb, xb, AcsA, prevB, Dv, Ybuf);

  // 8. gated RMSNorm
  k_norm<<<MTOK, 256, 0, stream>>>(Ybuf, zx, normw, yn);

  // 9. out_proj GEMM -> d_out (f32)
  dim3 g2(HID / 128, MTOK / 128);
  k_gemm_bf16<<<g2, 256, smg, stream>>>(yn, wout_b, out, DIN, HID);
}
